// DiffGlue_86809878986846
// MI455X (gfx1250) — compile-verified, matrix operand order checked by reading, not run on hardware
//
#include <hip/hip_runtime.h>
#include <hip/hip_bf16.h>

typedef _Float16 v16h __attribute__((ext_vector_type(16)));
typedef _Float16 h8   __attribute__((ext_vector_type(8)));
typedef float    v8f  __attribute__((ext_vector_type(8)));

static constexpr int BB = 4, MM = 2048, NNc = 2048, DDc = 256;
static constexpr float THRESH = 0.1f;
static constexpr float LNEPS = 1e-5f;

// ----------------------------------------------------------------------------
// elementwise f32 -> f16
// ----------------------------------------------------------------------------
__global__ __launch_bounds__(256) void k_f32_to_f16(const float* __restrict__ in,
                                                    _Float16* __restrict__ out,
                                                    long long n) {
  long long i = (long long)blockIdx.x * 256 + threadIdx.x;
  if (i < n) out[i] = (_Float16)in[i];
}

// per-batch transpose: in (R,C) -> out (C,R)
__global__ __launch_bounds__(256) void k_transpose_h(const _Float16* __restrict__ in,
                                                     _Float16* __restrict__ out,
                                                     int R, int C) {
  long long base = (long long)blockIdx.z * R * C;
  long long idx = (long long)blockIdx.x * 256 + threadIdx.x;
  if (idx < (long long)R * C) {
    int r = (int)(idx / C), c = (int)(idx % C);
    out[base + (long long)c * R + r] = in[base + idx];
  }
}

// fragment gather: halves [0,8) from base, [8,16) from base+16 halves
__device__ __forceinline__ v16h load_frag(const _Float16* base) {
  const h8* p = (const h8*)base;
  h8 lo = p[0];
  h8 hi = p[2];
  return __builtin_shufflevector(lo, hi, 0, 1, 2, 3, 4, 5, 6, 7,
                                 8, 9, 10, 11, 12, 13, 14, 15);
}

// ----------------------------------------------------------------------------
// Generic WMMA GEMM (NT): C[m,n] = scale*(sum_k A[m,k]*B[n,k] + bias[n])
//                                 (+ residual[m,n]) (+ beta*Cprev)
// f16 A/B fragments, f32 accumulate via v_wmma_f32_16x16x32_f16.
// Block: 256 threads = 8 waves, block tile 64 rows x 256 cols.
// Wave w: rows [rowW, rowW+32) as 2 A-frags, cols [colW, colW+64) as 4 B-frags,
// 8 accumulators (8 WMMA per 32-wide K-step).
// K-loop is unrolled 2x with ping-pong fragment sets so global loads write
// DIRECTLY into the WMMA operand registers (no rotation movs): load set1 @kk+32,
// compute set0, load set0 @kk+64, compute set1.  Requires K % 64 == 0 (all K
// used here are 256/512/2048).
// Fragment layouts per CDNA5 ISA 7.12.2 (wave32):
//   A 16x32 f16: lane L: row = L&15, group g=L>>4; halves e: k = e + 8g (+8 if e>=8)
//   B mirrored with column = L&15.
//   C/D 16x16 f32: lane L: col = L&15; vgpr r: row = r + 8*(L>>4)
// ----------------------------------------------------------------------------
__global__ __launch_bounds__(256)
void k_wmma_gemm_nt(const _Float16* __restrict__ A, int lda, long long sA,
                    const _Float16* __restrict__ Bm, int ldb, long long sB,
                    float* __restrict__ Cf, _Float16* __restrict__ Ch,
                    int ldc, long long sC,
                    const float* __restrict__ bias,
                    const float* __restrict__ res, int ldres,
                    float scale, int beta, int K) {
  const int lane = threadIdx.x & 31;
  const int wv = threadIdx.x >> 5;
  const int bz = blockIdx.z;
  const long long row0 = (long long)blockIdx.y * 64 + (wv >> 2) * 32;
  const long long col0 = (long long)blockIdx.x * 256 + (wv & 3) * 64;
  const int r16 = lane & 15, g = lane >> 4;

  const _Float16* Ap0 = A + bz * sA + (row0 + r16) * lda + 8 * g;
  const _Float16* Ap1 = Ap0 + 16LL * lda;
  const _Float16* Bp[4];
#pragma unroll
  for (int t = 0; t < 4; ++t)
    Bp[t] = Bm + bz * sB + (col0 + t * 16 + r16) * ldb + 8 * g;

  v8f acc[2][4] = {};
  v16h af0[2], bf0[4], af1[2], bf1[4];

  // prologue: fragment set 0 at kk = 0
  af0[0] = load_frag(Ap0);
  af0[1] = load_frag(Ap1);
#pragma unroll
  for (int t = 0; t < 4; ++t) bf0[t] = load_frag(Bp[t]);

  for (int kk = 0; kk < K; kk += 64) {
    // load set 1 at kk+32 (always in range: K % 64 == 0)
    af1[0] = load_frag(Ap0 + kk + 32);
    af1[1] = load_frag(Ap1 + kk + 32);
#pragma unroll
    for (int t = 0; t < 4; ++t) bf1[t] = load_frag(Bp[t] + kk + 32);

    // compute on set 0
#pragma unroll
    for (int rt = 0; rt < 2; ++rt)
#pragma unroll
      for (int t = 0; t < 4; ++t)
        acc[rt][t] = __builtin_amdgcn_wmma_f32_16x16x32_f16(
            false, af0[rt], false, bf0[t], (short)0, acc[rt][t], false, false);

    // load set 0 at kk+64
    if (kk + 64 < K) {
      af0[0] = load_frag(Ap0 + kk + 64);
      af0[1] = load_frag(Ap1 + kk + 64);
#pragma unroll
      for (int t = 0; t < 4; ++t) bf0[t] = load_frag(Bp[t] + kk + 64);
      __builtin_prefetch(Ap0 + kk + 96, 0, 1);
      __builtin_prefetch(Ap1 + kk + 96, 0, 1);
    }

    // compute on set 1
#pragma unroll
    for (int rt = 0; rt < 2; ++rt)
#pragma unroll
      for (int t = 0; t < 4; ++t)
        acc[rt][t] = __builtin_amdgcn_wmma_f32_16x16x32_f16(
            false, af1[rt], false, bf1[t], (short)0, acc[rt][t], false, false);
  }

  const int cn = lane & 15;
#pragma unroll
  for (int rt = 0; rt < 2; ++rt) {
#pragma unroll
    for (int t = 0; t < 4; ++t) {
      long long n = col0 + t * 16 + cn;
      float bb = bias ? bias[n] : 0.0f;
#pragma unroll
      for (int r = 0; r < 8; ++r) {
        long long m = row0 + rt * 16 + r + 8 * g;
        float v = (acc[rt][t][r] + bb) * scale;
        if (res) v += res[m * (long long)ldres + n];
        long long ci = bz * sC + m * (long long)ldc + n;
        if (Cf) {
          if (beta) v += Cf[ci];
          Cf[ci] = v;
        } else {
          Ch[ci] = (_Float16)v;
        }
      }
    }
  }
}

// ----------------------------------------------------------------------------
// adjacency softmaxes
// ----------------------------------------------------------------------------
__global__ __launch_bounds__(256)
void k_row_softmax(const float* __restrict__ adj, _Float16* __restrict__ out, int n) {
  __shared__ float red[256];
  const int tid = threadIdx.x;
  const float* a = adj + (long long)blockIdx.x * n;
  _Float16* o = out + (long long)blockIdx.x * n;
  float mx = -3.402823466e38f;
  for (int j = tid; j < n; j += 256) mx = fmaxf(mx, a[j]);
  red[tid] = mx; __syncthreads();
  for (int s = 128; s > 0; s >>= 1) { if (tid < s) red[tid] = fmaxf(red[tid], red[tid + s]); __syncthreads(); }
  mx = red[0]; __syncthreads();
  float sm = 0.f;
  for (int j = tid; j < n; j += 256) sm += expf(a[j] - mx);
  red[tid] = sm; __syncthreads();
  for (int s = 128; s > 0; s >>= 1) { if (tid < s) red[tid] += red[tid + s]; __syncthreads(); }
  float inv = 1.0f / red[0];
  for (int j = tid; j < n; j += 256) o[j] = (_Float16)(expf(a[j] - mx) * inv);
}

// softmax over axis-1 (columns), output TRANSPOSED: outT[b][n][m]
__global__ __launch_bounds__(256)
void k_col_softmax(const float* __restrict__ adj, _Float16* __restrict__ outT,
                   int Mr, int Nc) {
  __shared__ float red[256];
  const int tid = threadIdx.x;
  int b = blockIdx.x / Nc, n = blockIdx.x % Nc;
  const float* a = adj + (long long)b * Mr * Nc + n;
  _Float16* o = outT + (long long)blockIdx.x * Mr;
  float mx = -3.402823466e38f;
  for (int i = tid; i < Mr; i += 256) mx = fmaxf(mx, a[(long long)i * Nc]);
  red[tid] = mx; __syncthreads();
  for (int s = 128; s > 0; s >>= 1) { if (tid < s) red[tid] = fmaxf(red[tid], red[tid + s]); __syncthreads(); }
  mx = red[0]; __syncthreads();
  float sm = 0.f;
  for (int i = tid; i < Mr; i += 256) sm += expf(a[(long long)i * Nc] - mx);
  red[tid] = sm; __syncthreads();
  for (int s = 128; s > 0; s >>= 1) { if (tid < s) red[tid] += red[tid + s]; __syncthreads(); }
  float inv = 1.0f / red[0];
  for (int i = tid; i < Mr; i += 256) o[i] = (_Float16)(expf(a[(long long)i * Nc] - mx) * inv);
}

// ----------------------------------------------------------------------------
// LayerNorm (512 features) + exact GELU, f32 in, f16 out. Block per row.
// ----------------------------------------------------------------------------
__global__ __launch_bounds__(256)
void k_ln_gelu(const float* __restrict__ h, const float* __restrict__ gam,
               const float* __restrict__ bet, _Float16* __restrict__ out) {
  __shared__ float red[256];
  const int tid = threadIdx.x;
  const int F = 512;
  const float* hp = h + (long long)blockIdx.x * F;
  _Float16* op = out + (long long)blockIdx.x * F;
  float v0 = hp[tid], v1 = hp[tid + 256];
  red[tid] = v0 + v1; __syncthreads();
  for (int s = 128; s > 0; s >>= 1) { if (tid < s) red[tid] += red[tid + s]; __syncthreads(); }
  float mu = red[0] / F; __syncthreads();
  float d0 = v0 - mu, d1 = v1 - mu;
  red[tid] = d0 * d0 + d1 * d1; __syncthreads();
  for (int s = 128; s > 0; s >>= 1) { if (tid < s) red[tid] += red[tid + s]; __syncthreads(); }
  float inv = rsqrtf(red[0] / F + LNEPS);
  float y0 = d0 * inv * gam[tid] + bet[tid];
  float y1 = d1 * inv * gam[tid + 256] + bet[tid + 256];
  op[tid] = (_Float16)(0.5f * y0 * (1.0f + erff(y0 * 0.70710678118654752f)));
  op[tid + 256] = (_Float16)(0.5f * y1 * (1.0f + erff(y1 * 0.70710678118654752f)));
}

// ----------------------------------------------------------------------------
// z = d @ Wz + bz ; ls = log_sigmoid(z); lsn = log_sigmoid(-z). Warp per row.
// ----------------------------------------------------------------------------
__global__ __launch_bounds__(256)
void k_z_logsig(const _Float16* __restrict__ dmat, const float* __restrict__ Wz,
                const float* __restrict__ bz, float* __restrict__ ls,
                float* __restrict__ lsn, int rows, int Dd) {
  int w = (blockIdx.x * 256 + threadIdx.x) >> 5;
  int lane = threadIdx.x & 31;
  if (w >= rows) return;
  const _Float16* dp = dmat + (long long)w * Dd;
  float s = 0.f;
  for (int k = lane; k < Dd; k += 32) s += (float)dp[k] * Wz[k];
  for (int off = 16; off > 0; off >>= 1) s += __shfl_down(s, off, 32);
  if (lane == 0) {
    float z = s + bz[0];
    float l1p = log1pf(expf(-fabsf(z)));
    ls[w] = fminf(z, 0.f) - l1p;
    lsn[w] = fminf(-z, 0.f) - l1p;
  }
}

// ----------------------------------------------------------------------------
// logsumexp over rows / cols of sim
// ----------------------------------------------------------------------------
__global__ __launch_bounds__(256)
void k_row_lse(const float* __restrict__ sim, float* __restrict__ lse, int n) {
  __shared__ float red[256];
  const int tid = threadIdx.x;
  const float* row = sim + (long long)blockIdx.x * n;
  float mx = -3.402823466e38f;
  for (int j = tid; j < n; j += 256) mx = fmaxf(mx, row[j]);
  red[tid] = mx; __syncthreads();
  for (int s = 128; s > 0; s >>= 1) { if (tid < s) red[tid] = fmaxf(red[tid], red[tid + s]); __syncthreads(); }
  mx = red[0]; __syncthreads();
  float sm = 0.f;
  for (int j = tid; j < n; j += 256) sm += expf(row[j] - mx);
  red[tid] = sm; __syncthreads();
  for (int s = 128; s > 0; s >>= 1) { if (tid < s) red[tid] += red[tid + s]; __syncthreads(); }
  if (tid == 0) lse[blockIdx.x] = mx + logf(red[0]);
}

__global__ __launch_bounds__(256)
void k_col_lse(const float* __restrict__ sim, float* __restrict__ lse, int Mr, int Nc) {
  __shared__ float red[256];
  const int tid = threadIdx.x;
  int b = blockIdx.x / Nc, n = blockIdx.x % Nc;
  const float* col = sim + (long long)b * Mr * Nc + n;
  float mx = -3.402823466e38f;
  for (int i = tid; i < Mr; i += 256) mx = fmaxf(mx, col[(long long)i * Nc]);
  red[tid] = mx; __syncthreads();
  for (int s = 128; s > 0; s >>= 1) { if (tid < s) red[tid] = fmaxf(red[tid], red[tid + s]); __syncthreads(); }
  mx = red[0]; __syncthreads();
  float sm = 0.f;
  for (int i = tid; i < Mr; i += 256) sm += expf(col[(long long)i * Nc] - mx);
  red[tid] = sm; __syncthreads();
  for (int s = 128; s > 0; s >>= 1) { if (tid < s) red[tid] += red[tid + s]; __syncthreads(); }
  if (tid == 0) lse[blockIdx.x] = mx + logf(red[0]);
}

// ----------------------------------------------------------------------------
// scores assembly: (M+1)x(N+1), inner = 2*sim - lseR - lseC + ls0 + ls1
// ----------------------------------------------------------------------------
__global__ __launch_bounds__(256)
void k_scores(const float* __restrict__ sim, const float* __restrict__ lseR,
              const float* __restrict__ lseC, const float* __restrict__ ls0,
              const float* __restrict__ ls1, const float* __restrict__ lsn0,
              const float* __restrict__ lsn1, float* __restrict__ scores,
              int Mr, int Nc) {
  long long P = (long long)(Mr + 1) * (Nc + 1);
  long long idx = (long long)blockIdx.x * 256 + threadIdx.x;
  int b = blockIdx.y;
  if (idx >= P) return;
  int i = (int)(idx / (Nc + 1)), j = (int)(idx % (Nc + 1));
  float val;
  if (i < Mr && j < Nc) {
    val = 2.0f * sim[(long long)b * Mr * Nc + (long long)i * Nc + j]
          - lseR[b * Mr + i] - lseC[b * Nc + j] + ls0[b * Mr + i] + ls1[b * Nc + j];
  } else if (i < Mr) {
    val = lsn0[b * Mr + i];
  } else if (j < Nc) {
    val = lsn1[b * Nc + j];
  } else {
    val = 0.0f;
  }
  scores[(long long)b * P + idx] = val;
}

// ----------------------------------------------------------------------------
// argmax over inner block (first-max tie rule, like jnp.argmax)
// ----------------------------------------------------------------------------
__global__ __launch_bounds__(256)
void k_argmax_row(const float* __restrict__ scores, int Mr, int Nc,
                  float* __restrict__ mv, int* __restrict__ mi) {
  __shared__ float rv[256];
  __shared__ int ri[256];
  const int tid = threadIdx.x;
  long long P = (long long)(Mr + 1) * (Nc + 1);
  int b = blockIdx.x / Mr, m = blockIdx.x % Mr;
  const float* row = scores + (long long)b * P + (long long)m * (Nc + 1);
  float bv = -3.402823466e38f; int bi = 0x7fffffff;
  for (int j = tid; j < Nc; j += 256) {
    float v = row[j];
    if (v > bv || (v == bv && j < bi)) { bv = v; bi = j; }
  }
  rv[tid] = bv; ri[tid] = bi; __syncthreads();
  for (int s = 128; s > 0; s >>= 1) {
    if (tid < s) {
      if (rv[tid + s] > rv[tid] || (rv[tid + s] == rv[tid] && ri[tid + s] < ri[tid])) {
        rv[tid] = rv[tid + s]; ri[tid] = ri[tid + s];
      }
    }
    __syncthreads();
  }
  if (tid == 0) { mv[blockIdx.x] = rv[0]; mi[blockIdx.x] = ri[0]; }
}

__global__ __launch_bounds__(256)
void k_argmax_col(const float* __restrict__ scores, int Mr, int Nc,
                  float* __restrict__ mv, int* __restrict__ mi) {
  __shared__ float rv[256];
  __shared__ int ri[256];
  const int tid = threadIdx.x;
  long long P = (long long)(Mr + 1) * (Nc + 1);
  int b = blockIdx.x / Nc, n = blockIdx.x % Nc;
  const float* col = scores + (long long)b * P + n;
  float bv = -3.402823466e38f; int bi = 0x7fffffff;
  for (int i = tid; i < Mr; i += 256) {
    float v = col[(long long)i * (Nc + 1)];
    if (v > bv || (v == bv && i < bi)) { bv = v; bi = i; }
  }
  rv[tid] = bv; ri[tid] = bi; __syncthreads();
  for (int s = 128; s > 0; s >>= 1) {
    if (tid < s) {
      if (rv[tid + s] > rv[tid] || (rv[tid + s] == rv[tid] && ri[tid + s] < ri[tid])) {
        rv[tid] = rv[tid + s]; ri[tid] = ri[tid + s];
      }
    }
    __syncthreads();
  }
  if (tid == 0) { mv[blockIdx.x] = rv[0]; mi[blockIdx.x] = ri[0]; }
}

// ----------------------------------------------------------------------------
// mutual-match filtering
// ----------------------------------------------------------------------------
__global__ __launch_bounds__(256)
void k_filter0(const int* __restrict__ i0, const int* __restrict__ i1,
               const float* __restrict__ max0v, float* __restrict__ out_m0,
               float* __restrict__ out_ms0, int Mr, int Nc) {
  int idx = blockIdx.x * 256 + threadIdx.x;
  if (idx >= BB * Mr) return;
  int b = idx / Mr, m = idx % Mr;
  int j = i0[idx];
  bool mutual = (i1[b * Nc + j] == m);
  float ms = mutual ? expf(max0v[idx]) : 0.0f;
  bool valid = mutual && (ms > THRESH);
  out_m0[idx] = valid ? (float)j : -1.0f;
  out_ms0[idx] = ms;
}

__global__ __launch_bounds__(256)
void k_filter1(const int* __restrict__ i0, const int* __restrict__ i1,
               const float* __restrict__ out_m0, const float* __restrict__ out_ms0,
               float* __restrict__ out_m1, float* __restrict__ out_ms1,
               int Mr, int Nc) {
  int idx = blockIdx.x * 256 + threadIdx.x;
  if (idx >= BB * Nc) return;
  int b = idx / Nc, n = idx % Nc;
  int i = i1[idx];
  bool mutual = (i0[b * Mr + i] == n);
  float ms = mutual ? out_ms0[b * Mr + i] : 0.0f;
  bool valid = mutual && (out_m0[b * Mr + i] >= 0.0f);
  out_m1[idx] = valid ? (float)i : -1.0f;
  out_ms1[idx] = ms;
}

// ----------------------------------------------------------------------------
// host
// ----------------------------------------------------------------------------
static inline void launch_gemm(hipStream_t s, const _Float16* A, int lda, long long sA,
                               const _Float16* B, int ldb, long long sB,
                               float* Cf, _Float16* Ch, int ldc, long long sC,
                               const float* bias, const float* res, int ldres,
                               float scale, int beta, int rows, int cols, int K, int batch) {
  dim3 g(cols / 256, rows / 64, batch);
  k_wmma_gemm_nt<<<g, dim3(256), 0, s>>>(A, lda, sA, B, ldb, sB, Cf, Ch, ldc, sC,
                                         bias, res, ldres, scale, beta, K);
}

extern "C" void kernel_launch(void* const* d_in, const int* in_sizes, int n_in,
                              void* d_out, int out_size, void* d_ws, size_t ws_size,
                              hipStream_t stream) {
  const float* x0 = (const float*)d_in[0];
  const float* x1 = (const float*)d_in[1];
  const float* adj = (const float*)d_in[2];
  const float* Wv = (const float*)d_in[3];
  const float* bv = (const float*)d_in[4];
  const float* Wo = (const float*)d_in[5];
  const float* bo = (const float*)d_in[6];
  const float* Wf1 = (const float*)d_in[7];
  const float* bf1 = (const float*)d_in[8];
  const float* ln_g = (const float*)d_in[9];
  const float* ln_b = (const float*)d_in[10];
  const float* Wf2 = (const float*)d_in[11];
  const float* bf2 = (const float*)d_in[12];
  const float* Wfp = (const float*)d_in[13];
  const float* bfp = (const float*)d_in[14];
  const float* Wz = (const float*)d_in[15];
  const float* bz = (const float*)d_in[16];

  const long long R = (long long)BB * MM;       // 8192 flattened rows
  char* w = (char*)d_ws;
  size_t off = 0;
  auto alloc = [&](size_t bytes) -> void* {
    off = (off + 255) & ~(size_t)255;
    void* p = w + off;
    off += bytes;
    return p;
  };

  _Float16* x0h  = (_Float16*)alloc(R * DDc * 2);
  _Float16* x1h  = (_Float16*)alloc(R * DDc * 2);
  _Float16* Wvh  = (_Float16*)alloc(DDc * DDc * 2);
  _Float16* Woh  = (_Float16*)alloc(DDc * DDc * 2);
  _Float16* Wf1h = (_Float16*)alloc(512 * 512 * 2);
  _Float16* Wf2h = (_Float16*)alloc(256 * 512 * 2);
  _Float16* Wfph = (_Float16*)alloc(DDc * DDc * 2);
  _Float16* v0h  = (_Float16*)alloc(R * DDc * 2);
  _Float16* v1h  = (_Float16*)alloc(R * DDc * 2);
  _Float16* v0t  = (_Float16*)alloc(R * DDc * 2);
  _Float16* v1t  = (_Float16*)alloc(R * DDc * 2);
  _Float16* A01h = (_Float16*)alloc((long long)BB * MM * NNc * 2);   // 32 MB
  _Float16* A10t = (_Float16*)alloc((long long)BB * MM * NNc * 2);   // 32 MB
  float*    sim  = (float*)A01h;   // alias: A01h+A10t (64 MB) reused after use
  _Float16* m0p  = (_Float16*)alloc(R * DDc * 2);
  _Float16* m1p  = (_Float16*)alloc(R * DDc * 2);
  _Float16* m0h  = (_Float16*)alloc(R * DDc * 2);
  _Float16* m1h  = (_Float16*)alloc(R * DDc * 2);
  float*    hbuf = (float*)alloc(R * 512 * 4);
  _Float16* hgel = (_Float16*)alloc(R * 512 * 2);
  _Float16* d0h  = (_Float16*)alloc(R * DDc * 2);
  _Float16* d1h  = (_Float16*)alloc(R * DDc * 2);
  _Float16* md0h = (_Float16*)alloc(R * DDc * 2);
  _Float16* md1h = (_Float16*)alloc(R * DDc * 2);
  float* lseR = (float*)alloc(R * 4);
  float* lseC = (float*)alloc(R * 4);
  float* ls0  = (float*)alloc(R * 4);
  float* ls1  = (float*)alloc(R * 4);
  float* lsn0 = (float*)alloc(R * 4);
  float* lsn1 = (float*)alloc(R * 4);
  int*   i0   = (int*)alloc(R * 4);
  int*   i1   = (int*)alloc(R * 4);
  float* max0v = (float*)alloc(R * 4);
  float* max1v = (float*)alloc(R * 4);

  float* out_scores = (float*)d_out;
  float* out_m0  = out_scores + (size_t)BB * (MM + 1) * (NNc + 1);
  float* out_m1  = out_m0 + (size_t)BB * MM;
  float* out_ms0 = out_m1 + (size_t)BB * NNc;
  float* out_ms1 = out_ms0 + (size_t)BB * MM;

  auto cvt = [&](const float* src, _Float16* dst, long long n) {
    k_f32_to_f16<<<dim3((unsigned)((n + 255) / 256)), dim3(256), 0, stream>>>(src, dst, n);
  };
  cvt(x0, x0h, R * DDc);
  cvt(x1, x1h, R * DDc);
  cvt(Wv, Wvh, DDc * DDc);
  cvt(Wo, Woh, DDc * DDc);
  cvt(Wf1, Wf1h, 512 * 512);
  cvt(Wf2, Wf2h, 256 * 512);
  cvt(Wfp, Wfph, DDc * DDc);

  // v = x @ Wv^T + bv  (f16 out)
  launch_gemm(stream, x0h, DDc, 0, Wvh, DDc, 0, nullptr, v0h, DDc, 0, bv, nullptr, 0,
              1.0f, 0, (int)R, DDc, DDc, 1);
  launch_gemm(stream, x1h, DDc, 0, Wvh, DDc, 0, nullptr, v1h, DDc, 0, bv, nullptr, 0,
              1.0f, 0, (int)R, DDc, DDc, 1);
  // v transposed per batch: (M,D) -> (D,M)
  {
    dim3 g((unsigned)((MM * DDc + 255) / 256), 1, BB);
    k_transpose_h<<<g, dim3(256), 0, stream>>>(v0h, v0t, MM, DDc);
    k_transpose_h<<<g, dim3(256), 0, stream>>>(v1h, v1t, NNc, DDc);
  }

  // adjacency softmaxes
  k_row_softmax<<<dim3(BB * MM), dim3(256), 0, stream>>>(adj, A01h, NNc);
  k_col_softmax<<<dim3(BB * NNc), dim3(256), 0, stream>>>(adj, A10t, MM, NNc);

  // message aggregation (batched): m0p = A01 @ v1, m1p = A10^T @ v0
  launch_gemm(stream, A01h, NNc, (long long)MM * NNc, v1t, NNc, (long long)DDc * NNc,
              nullptr, m0p, DDc, (long long)MM * DDc, nullptr, nullptr, 0,
              1.0f, 0, MM, DDc, NNc, BB);
  launch_gemm(stream, A10t, MM, (long long)MM * NNc, v0t, MM, (long long)DDc * MM,
              nullptr, m1p, DDc, (long long)NNc * DDc, nullptr, nullptr, 0,
              1.0f, 0, NNc, DDc, MM, BB);

  // m = mp @ Wo^T + bo
  launch_gemm(stream, m0p, DDc, 0, Woh, DDc, 0, nullptr, m0h, DDc, 0, bo, nullptr, 0,
              1.0f, 0, (int)R, DDc, DDc, 1);
  launch_gemm(stream, m1p, DDc, 0, Woh, DDc, 0, nullptr, m1h, DDc, 0, bo, nullptr, 0,
              1.0f, 0, (int)R, DDc, DDc, 1);

  // FFN per side: h = cat(x,m) @ Wf1^T + bf1 -> LN -> gelu -> d = x + h @ Wf2^T + bf2
  for (int side = 0; side < 2; ++side) {
    const _Float16* xh = side ? x1h : x0h;
    const _Float16* mh = side ? m1h : m0h;
    const float* xres = side ? x1 : x0;
    _Float16* dh = side ? d1h : d0h;
    launch_gemm(stream, xh, DDc, 0, Wf1h, 512, 0, hbuf, nullptr, 512, 0,
                nullptr, nullptr, 0, 1.0f, 0, (int)R, 512, DDc, 1);
    launch_gemm(stream, mh, DDc, 0, Wf1h + 256, 512, 0, hbuf, nullptr, 512, 0,
                bf1, nullptr, 0, 1.0f, 1, (int)R, 512, DDc, 1);
    k_ln_gelu<<<dim3((unsigned)R), dim3(256), 0, stream>>>(hbuf, ln_g, ln_b, hgel);
    launch_gemm(stream, hgel, 512, 0, Wf2h, 512, 0, nullptr, dh, DDc, 0,
                bf2, xres, DDc, 1.0f, 0, (int)R, DDc, 512, 1);
  }

  // md = (d @ Wfp^T + bfp) / D^0.25   (D^0.25 = 4)
  launch_gemm(stream, d0h, DDc, 0, Wfph, DDc, 0, nullptr, md0h, DDc, 0, bfp, nullptr, 0,
              0.25f, 0, (int)R, DDc, DDc, 1);
  launch_gemm(stream, d1h, DDc, 0, Wfph, DDc, 0, nullptr, md1h, DDc, 0, bfp, nullptr, 0,
              0.25f, 0, (int)R, DDc, DDc, 1);

  // sim = md0 @ md1^T (batched, f32 out, 64MB aliased over A01h/A10t)
  launch_gemm(stream, md0h, DDc, (long long)MM * DDc, md1h, DDc, (long long)NNc * DDc,
              sim, nullptr, NNc, (long long)MM * NNc, nullptr, nullptr, 0,
              1.0f, 0, MM, NNc, DDc, BB);

  // z / log-sigmoid terms
  k_z_logsig<<<dim3((unsigned)(R / 8)), dim3(256), 0, stream>>>(d0h, Wz, bz, ls0, lsn0, (int)R, DDc);
  k_z_logsig<<<dim3((unsigned)(R / 8)), dim3(256), 0, stream>>>(d1h, Wz, bz, ls1, lsn1, (int)R, DDc);

  // logsumexp reductions
  k_row_lse<<<dim3(BB * MM), dim3(256), 0, stream>>>(sim, lseR, NNc);
  k_col_lse<<<dim3(BB * NNc), dim3(256), 0, stream>>>(sim, lseC, MM, NNc);

  // scores
  {
    long long P = (long long)(MM + 1) * (NNc + 1);
    dim3 g((unsigned)((P + 255) / 256), BB);
    k_scores<<<g, dim3(256), 0, stream>>>(sim, lseR, lseC, ls0, ls1, lsn0, lsn1,
                                          out_scores, MM, NNc);
  }

  // match filtering
  k_argmax_row<<<dim3(BB * MM), dim3(256), 0, stream>>>(out_scores, MM, NNc, max0v, i0);
  k_argmax_col<<<dim3(BB * NNc), dim3(256), 0, stream>>>(out_scores, MM, NNc, max1v, i1);
  k_filter0<<<dim3((unsigned)((BB * MM + 255) / 256)), dim3(256), 0, stream>>>(
      i0, i1, max0v, out_m0, out_ms0, MM, NNc);
  k_filter1<<<dim3((unsigned)((BB * NNc + 255) / 256)), dim3(256), 0, stream>>>(
      i0, i1, out_m0, out_ms0, out_m1, out_ms1, MM, NNc);
}
